// TritonMatryoshkaLinear_59820304498924
// MI455X (gfx1250) — matrix operand order repaired and checked by reading, not compiled, not run on hardware
//
#include <hip/hip_runtime.h>

// ---------------------------------------------------------------------------
// Matryoshka staircase block-sparse GEMM for MI455X (gfx1250).
//
// out[:, o0:o1] (level L) = x[:, :kin_L] @ W[:kin_L, o0:o1]
// levels: (kin, ocols) = (1024, 0:1024), (2048, 1024:2048), (4096, 2048:4096)
//
// Precision strategy: split-bf16 emulation of fp32 GEMM.
//   f = hi(bf16) + lo(bf16);  A*B ~= Ah*Bh + Ah*Bl + Al*Bh  (fp32 accumulate)
// 3x v_wmma_f32_16x16x32_bf16 per 32-wide k-chunk -> ~3/8 of bf16 matrix peak,
// far faster than the native 16x16x4 f32 WMMA while keeping ~1e-5 relative
// accuracy (inputs carry 17 effective mantissa bits).
// ---------------------------------------------------------------------------

typedef __bf16 v16bf __attribute__((ext_vector_type(16)));
typedef __bf16 v8bf  __attribute__((ext_vector_type(8)));
typedef __bf16 v4bf  __attribute__((ext_vector_type(4)));
typedef float  v8f   __attribute__((ext_vector_type(8)));

#define M_TOK 8192
#define KDIM  4096
#define NDIM  4096
#define BM    128
#define BN    128
#define BK    32
#define LDA   40   // padded LDS row stride in bf16 elements (80B, 16B-aligned)

union FragBF {
    v16bf v;
    v8bf  h[2];
};

struct BF2 { __bf16 hi; __bf16 lo; };

__device__ __forceinline__ BF2 split_bf16(float f) {
    BF2 r;
    r.hi = (__bf16)f;                    // RTNE
    r.lo = (__bf16)(f - (float)r.hi);    // residual, next 8 mantissa bits
    return r;
}

__global__ __launch_bounds__(256)
void matryoshka_wmma_kernel(const float* __restrict__ x,
                            const float* __restrict__ W,
                            float* __restrict__ out) {
    __shared__ __bf16 Ah[BM * LDA];
    __shared__ __bf16 Al[BM * LDA];
    __shared__ __bf16 Bh[BN * LDA];   // stored transposed: [n][k]
    __shared__ __bf16 Bl[BN * LDA];

    const int tid   = threadIdx.x;     // 0..255, 8 waves of 32
    const int lane  = tid & 31;
    const int wave  = tid >> 5;
    const int wm    = wave >> 2;       // 0..1 : 64-row slab
    const int wn    = wave & 3;        // 0..3 : 32-col slab
    const int lm    = lane & 15;       // M/N index within 16x16 tile
    const int hm    = lane >> 4;       // half-wave selector (K-offset group)

    const int m0    = blockIdx.y * BM;
    const int ntile = blockIdx.x;
    const int n0    = ntile * BN;
    // staircase level -> K extent (128-col tiles: 0..7 -> 1024, 8..15 -> 2048, rest -> 4096)
    const int Kmax  = (ntile < 8) ? 1024 : (ntile < 16) ? 2048 : 4096;

    v8f acc[4][2];
#pragma unroll
    for (int mi = 0; mi < 4; ++mi)
#pragma unroll
        for (int ni = 0; ni < 2; ++ni)
            acc[mi][ni] = (v8f){0.f, 0.f, 0.f, 0.f, 0.f, 0.f, 0.f, 0.f};

    for (int k0 = 0; k0 < Kmax; k0 += BK) {
        // ---- stage X tile (BM x BK fp32) as hi/lo bf16, row-major -----------
#pragma unroll
        for (int i = 0; i < 4; ++i) {
            const int idx = tid + i * 256;      // 0..1023 float4 slots
            const int r   = idx >> 3;           // row 0..127
            const int cg  = idx & 7;            // 4-col group 0..7
            const float4 v = *(const float4*)(x + (size_t)(m0 + r) * KDIM + (k0 + cg * 4));
            const BF2 s0 = split_bf16(v.x);
            const BF2 s1 = split_bf16(v.y);
            const BF2 s2 = split_bf16(v.z);
            const BF2 s3 = split_bf16(v.w);
            v4bf h, l;
            h[0] = s0.hi; h[1] = s1.hi; h[2] = s2.hi; h[3] = s3.hi;
            l[0] = s0.lo; l[1] = s1.lo; l[2] = s2.lo; l[3] = s3.lo;
            *(v4bf*)(&Ah[r * LDA + cg * 4]) = h;
            *(v4bf*)(&Al[r * LDA + cg * 4]) = l;
        }
        // ---- stage W tile (BK x BN fp32) transposed to [n][k] hi/lo bf16 ----
#pragma unroll
        for (int i = 0; i < 4; ++i) {
            const int idx = tid + i * 256;
            const int r   = idx >> 5;           // k row 0..31
            const int cg  = idx & 31;           // 4-col group 0..31
            const float4 v = *(const float4*)(W + (size_t)(k0 + r) * NDIM + (n0 + cg * 4));
            const float f[4] = {v.x, v.y, v.z, v.w};
#pragma unroll
            for (int e = 0; e < 4; ++e) {
                const BF2 s = split_bf16(f[e]);
                Bh[(cg * 4 + e) * LDA + r] = s.hi;
                Bl[(cg * 4 + e) * LDA + r] = s.lo;
            }
        }
        // prefetch next k-tile into cache (gfx1250 global_prefetch_b8)
        if (k0 + BK < Kmax) {
            const int r  = tid >> 3, cg  = tid & 7;
            __builtin_prefetch(x + (size_t)(m0 + r) * KDIM + (k0 + BK + cg * 4), 0, 1);
            const int r2 = tid >> 5, cg2 = tid & 31;
            __builtin_prefetch(W + (size_t)(k0 + BK + r2) * NDIM + (n0 + cg2 * 4), 0, 1);
        }
        __syncthreads();

        // ---- build WMMA fragments from LDS ---------------------------------
        // A 16x32 bf16 layout: lanes 0-15 row M, elems 0..7 = K[hm*8 .. +7],
        //                      elems 8..15 = K[16+hm*8 .. +7]
        FragBF afh[4], afl[4];
#pragma unroll
        for (int mi = 0; mi < 4; ++mi) {
            const int row = wm * 64 + mi * 16 + lm;
            const __bf16* bh = &Ah[row * LDA];
            const __bf16* bl = &Al[row * LDA];
            afh[mi].h[0] = *(const v8bf*)(bh + hm * 8);
            afh[mi].h[1] = *(const v8bf*)(bh + 16 + hm * 8);
            afl[mi].h[0] = *(const v8bf*)(bl + hm * 8);
            afl[mi].h[1] = *(const v8bf*)(bl + 16 + hm * 8);
        }
        // B 32x16 bf16 layout: lane n elems 0..7 = K[hm*8 .. +7], elems 8..15 = K[16+hm*8 .. +7]
        FragBF bfh[2], bfl[2];
#pragma unroll
        for (int ni = 0; ni < 2; ++ni) {
            const int col = wn * 32 + ni * 16 + lm;
            const __bf16* bh = &Bh[col * LDA + hm * 8];
            const __bf16* bl = &Bl[col * LDA + hm * 8];
            bfh[ni].h[0] = *(const v8bf*)(bh);
            bfh[ni].h[1] = *(const v8bf*)(bh + 16);
            bfl[ni].h[0] = *(const v8bf*)(bl);
            bfl[ni].h[1] = *(const v8bf*)(bl + 16);
        }

        // ---- 24 WMMAs: Ah*Bh + Ah*Bl + Al*Bh, fp32 accumulate --------------
#pragma unroll
        for (int mi = 0; mi < 4; ++mi)
#pragma unroll
            for (int ni = 0; ni < 2; ++ni) {
                acc[mi][ni] = __builtin_amdgcn_wmma_f32_16x16x32_bf16(
                    false, afh[mi].v, false, bfh[ni].v, (short)0, acc[mi][ni], false, false);
                acc[mi][ni] = __builtin_amdgcn_wmma_f32_16x16x32_bf16(
                    false, afh[mi].v, false, bfl[ni].v, (short)0, acc[mi][ni], false, false);
                acc[mi][ni] = __builtin_amdgcn_wmma_f32_16x16x32_bf16(
                    false, afl[mi].v, false, bfh[ni].v, (short)0, acc[mi][ni], false, false);
            }
        __syncthreads();
    }

    // ---- epilogue: C/D layout is M = vgpr + 8*lane[4], N = lane[3:0] -------
#pragma unroll
    for (int mi = 0; mi < 4; ++mi)
#pragma unroll
        for (int ni = 0; ni < 2; ++ni) {
            const int cn = n0 + wn * 32 + ni * 16 + lm;
            const int rm = m0 + wm * 64 + mi * 16 + hm * 8;
#pragma unroll
            for (int r = 0; r < 8; ++r)
                out[(size_t)(rm + r) * NDIM + cn] = acc[mi][ni][r];
        }
}

extern "C" void kernel_launch(void* const* d_in, const int* in_sizes, int n_in,
                              void* d_out, int out_size, void* d_ws, size_t ws_size,
                              hipStream_t stream) {
    (void)in_sizes; (void)n_in; (void)out_size; (void)d_ws; (void)ws_size;
    const float* x = (const float*)d_in[0];
    const float* W = (const float*)d_in[1];
    float* out     = (float*)d_out;

    dim3 grid(NDIM / BN, M_TOK / BM);   // 32 x 64 tiles of 128x128
    dim3 block(256);                    // 8 wave32 waves
    matryoshka_wmma_kernel<<<grid, block, 0, stream>>>(x, W, out);
}
